// RNN_73521250173478
// MI455X (gfx1250) — matrix operand order repaired and checked by reading, not compiled, not run on hardware
//
#include <hip/hip_runtime.h>
#include <hip/hip_bf16.h>
#include <cmath>

typedef __attribute__((ext_vector_type(16))) _Float16 v16h;
typedef __attribute__((ext_vector_type(8)))  _Float16 v8h;
typedef __attribute__((ext_vector_type(8)))  float    v8f;
typedef __attribute__((ext_vector_type(8)))  int      v8i_;

#define RNN_H 512
#define RNN_V 256
#define RNN_B 64
#define RNN_S 1024

#define WL_STRIDE 264   // 256 + 8 pad (halfs): 528B row stride -> banks rotate by 4
#define HB_STRIDE 520   // 512 + 8 pad (halfs): 1040B row stride -> banks rotate by 4

static __device__ __forceinline__ v16h cat8(v8h lo, v8h hi) {
    return __builtin_shufflevector(lo, hi, 0,1,2,3,4,5,6,7,8,9,10,11,12,13,14,15);
}

// ---------------------------------------------------------------------------
// Prep: W_hh [K=512][N=512] f32 -> WhhT [N=512][K=512] f16
//       W_hy [K=512][N=256] f32 -> WhyT [N=256][K=512] f16
// ---------------------------------------------------------------------------
__global__ void rnn_prep_weights(const float* __restrict__ W_hh,
                                 const float* __restrict__ W_hy,
                                 _Float16* __restrict__ WhhT,
                                 _Float16* __restrict__ WhyT) {
    int i = blockIdx.x * 256 + threadIdx.x;           // 0 .. 262143
    {   // 512x512 transpose-convert
        int n = i & (RNN_H - 1);
        int k = i >> 9;
        WhhT[n * RNN_H + k] = (_Float16)W_hh[k * RNN_H + n];
    }
    if (i < RNN_V * RNN_H) {                          // 256x512
        int n = i % RNN_V;
        int k = i / RNN_V;
        WhyT[n * RNN_H + k] = (_Float16)W_hy[k * RNN_V + n];
    }
}

// ---------------------------------------------------------------------------
// Kernel A: serial recurrence. 4 WGs x 512 threads (16 waves); LDS (296 KB)
// already limits to 1 WG/WGP, so 16 waves -> 4 waves/SIMD -> 256 VGPRs/wave.
// Wave w owns TWO 16-col strips: cols [32w,32w+16) and [32w+16,32w+32).
//   K 0..255   of W_hh^T: persistent in 2x64 VGPRs/wave (loaded once).
//   K 256..511 of W_hh^T: staged once into LDS (264 KB, bank-padded).
//   h state: f16 double buffer in LDS (padded).
//   Each A fragment feeds both strips' WMMAs (2 indep. acc chains per wave).
//   Token + W_ih gather software-pipelined one/two steps ahead.
// ---------------------------------------------------------------------------
__global__ __launch_bounds__(512) void rnn_recurrence(
        const int*      __restrict__ x,     // [64][1024]
        const float*    __restrict__ W_ih,  // [256][512]
        const float*    __restrict__ b_h,   // [512]
        const _Float16* __restrict__ WhhT,  // [512][512] (n-major)
        _Float16*       __restrict__ h16)   // [1024][64][512]
{
    __shared__ _Float16 WL[RNN_H * WL_STRIDE];        // 264 KB: k=256..511, n-major
    __shared__ _Float16 hbuf[2][16 * HB_STRIDE];      // 32.5 KB double buffer

    const int tid  = threadIdx.x;
    const int wave = tid >> 5;                        // 0..15
    const int lane = tid & 31;
    const int lrow = lane & 15;
    const int hi   = lane >> 4;                       // 0 or 1
    const int g    = blockIdx.x;                      // batch group 0..3
    const int nc0  = wave * 32 + lrow;                // strip-0 column
    const int nc1  = nc0 + 16;                        // strip-1 column

    // ---- one-time: stage upper K half of W_hh^T into LDS (tid == row) ----
    {
        const _Float16* src = WhhT + (size_t)tid * RNN_H + 256;
        _Float16*       dst = WL + tid * WL_STRIDE;
        #pragma unroll
        for (int j = 0; j < 256; j += 8)
            *(v8h*)(dst + j) = *(const v8h*)(src + j);
    }
    // h_{-1} = 0
    for (int i = tid; i < 16 * HB_STRIDE; i += 512)
        hbuf[0][i] = (_Float16)0.0f;

    // ---- persistent B fragments, k = 0..255, both strips (128 VGPRs) ----
    const _Float16* wrow0 = WhhT + (size_t)nc0 * RNN_H;
    const _Float16* wrow1 = WhhT + (size_t)nc1 * RNN_H;
    v16h B0[8], B1[8];
    #pragma unroll
    for (int kt = 0; kt < 8; ++kt) {
        B0[kt] = cat8(*(const v8h*)(wrow0 + kt * 32 + hi * 16),
                      *(const v8h*)(wrow0 + kt * 32 + hi * 16 + 8));
        B1[kt] = cat8(*(const v8h*)(wrow1 + kt * 32 + hi * 16),
                      *(const v8h*)(wrow1 + kt * 32 + hi * 16 + 8));
    }

    const float bias0 = b_h[nc0];
    const float bias1 = b_h[nc1];
    const _Float16* wl0 = WL + nc0 * WL_STRIDE;
    const _Float16* wl1 = WL + nc1 * WL_STRIDE;

    // ---- gather pipeline prologue ----
    v8f  xt0, xt1;      // W_ih rows for step t, strip 0/1 columns
    v8i_ tokv;          // tokens for step t+1
    #pragma unroll
    for (int r = 0; r < 8; ++r) {
        int m = r + hi * 8;
        int tok0 = x[(g * 16 + m) * RNN_S + 0];
        xt0[r]  = W_ih[tok0 * RNN_H + nc0];
        xt1[r]  = W_ih[tok0 * RNN_H + nc1];
        tokv[r] = x[(g * 16 + m) * RNN_S + 1];
    }

    __syncthreads();

    for (int t = 0; t < RNN_S; ++t) {
        // C init from the prefetched gather (loads completed last step)
        v8f acc0, acc1;
        #pragma unroll
        for (int r = 0; r < 8; ++r) {
            acc0[r] = xt0[r] + bias0;
            acc1[r] = xt1[r] + bias1;
        }

        // re-issue the pipeline: W_ih gather for t+1 (tokens resident),
        // tokens for t+2. Waits land at the top of the next step.
        {
            const int t2 = (t + 2) & (RNN_S - 1);
            #pragma unroll
            for (int r = 0; r < 8; ++r) {
                int m = r + hi * 8;
                xt0[r]  = W_ih[tokv[r] * RNN_H + nc0];
                xt1[r]  = W_ih[tokv[r] * RNN_H + nc1];
                tokv[r] = x[(g * 16 + m) * RNN_S + t2];
            }
        }

        const _Float16* hb = hbuf[t & 1];

        // k = 0..255: B from registers, shared A from LDS
        #pragma unroll
        for (int kt = 0; kt < 8; ++kt) {
            const int kbase = kt * 32;
            v16h a = cat8(*(const v8h*)(hb + lrow * HB_STRIDE + kbase + hi * 16),
                          *(const v8h*)(hb + lrow * HB_STRIDE + kbase + hi * 16 + 8));
            acc0 = __builtin_amdgcn_wmma_f32_16x16x32_f16(
                       false, a, false, B0[kt], (short)0, acc0, false, false);
            acc1 = __builtin_amdgcn_wmma_f32_16x16x32_f16(
                       false, a, false, B1[kt], (short)0, acc1, false, false);
        }
        // k = 256..511: B from LDS, shared A from LDS
        #pragma unroll
        for (int kt = 0; kt < 8; ++kt) {
            const int kbase = 256 + kt * 32;
            v16h a = cat8(*(const v8h*)(hb + lrow * HB_STRIDE + kbase + hi * 16),
                          *(const v8h*)(hb + lrow * HB_STRIDE + kbase + hi * 16 + 8));
            v16h b0 = cat8(*(const v8h*)(wl0 + kt * 32 + hi * 16),
                           *(const v8h*)(wl0 + kt * 32 + hi * 16 + 8));
            v16h b1 = cat8(*(const v8h*)(wl1 + kt * 32 + hi * 16),
                           *(const v8h*)(wl1 + kt * 32 + hi * 16 + 8));
            acc0 = __builtin_amdgcn_wmma_f32_16x16x32_f16(
                       false, a, false, b0, (short)0, acc0, false, false);
            acc1 = __builtin_amdgcn_wmma_f32_16x16x32_f16(
                       false, a, false, b1, (short)0, acc1, false, false);
        }

        // h = tanh(acc); publish both strips to other LDS buffer + global f16
        _Float16* hw = hbuf[(t + 1) & 1];
        _Float16* hg = h16 + ((size_t)t * RNN_B + g * 16) * RNN_H;
        #pragma unroll
        for (int r = 0; r < 8; ++r) {
            int m = r + hi * 8;
            _Float16 h0 = (_Float16)tanhf(acc0[r]);
            _Float16 h1 = (_Float16)tanhf(acc1[r]);
            hw[m * HB_STRIDE + nc0] = h0;
            hw[m * HB_STRIDE + nc1] = h1;
            hg[m * RNN_H + nc0]     = h0;
            hg[m * RNN_H + nc1]     = h1;
        }
        __syncthreads();
    }
}

// ---------------------------------------------------------------------------
// Kernel B: y = h @ W_hy + b_y as one big GEMM, M=65536, K=512, N=256.
// One 16x64 output tile per wave (4 accumulators), K-loop of 16 x 4 WMMAs.
// ---------------------------------------------------------------------------
__global__ __launch_bounds__(256) void rnn_output(
        const _Float16* __restrict__ h16,   // [65536][512]
        const _Float16* __restrict__ WhyT,  // [256][512] (n-major)
        const float*    __restrict__ b_y,   // [256]
        float*          __restrict__ y)     // [65536][256]
{
    const int wave = threadIdx.x >> 5;
    const int lane = threadIdx.x & 31;
    const int lrow = lane & 15;
    const int hi   = lane >> 4;
    const int tile = blockIdx.x * 8 + wave;   // 0..16383
    const int mt   = tile >> 2;               // m-tile (16 rows of S*B)
    const int ng   = tile & 3;                // 64-col group of V

    const _Float16* arow = h16 + ((size_t)mt * 16 + lrow) * RNN_H;

    v8f acc[4];
    #pragma unroll
    for (int nt = 0; nt < 4; ++nt) {
        float bb = b_y[ng * 64 + nt * 16 + lrow];
        #pragma unroll
        for (int r = 0; r < 8; ++r) acc[nt][r] = bb;
    }

    for (int kt = 0; kt < 16; ++kt) {
        const int kbase = kt * 32;
        v8h a0 = *(const v8h*)(arow + kbase + hi * 16);
        v8h a1 = *(const v8h*)(arow + kbase + hi * 16 + 8);
        v16h a = cat8(a0, a1);
        #pragma unroll
        for (int nt = 0; nt < 4; ++nt) {
            const _Float16* brow =
                WhyT + (size_t)(ng * 64 + nt * 16 + lrow) * RNN_H + kbase + hi * 16;
            v8h b0 = *(const v8h*)(brow);
            v8h b1 = *(const v8h*)(brow + 8);
            acc[nt] = __builtin_amdgcn_wmma_f32_16x16x32_f16(
                          false, a, false, cat8(b0, b1),
                          (short)0, acc[nt], false, false);
        }
    }

    float* out = y + (size_t)mt * 16 * RNN_V + ng * 64;
    #pragma unroll
    for (int nt = 0; nt < 4; ++nt)
        #pragma unroll
        for (int r = 0; r < 8; ++r)
            out[(r + hi * 8) * RNN_V + nt * 16 + lrow] = acc[nt][r];
}

// ---------------------------------------------------------------------------
extern "C" void kernel_launch(void* const* d_in, const int* in_sizes, int n_in,
                              void* d_out, int out_size, void* d_ws, size_t ws_size,
                              hipStream_t stream) {
    (void)in_sizes; (void)n_in; (void)out_size; (void)ws_size;

    const int*   x    = (const int*)  d_in[0];
    const float* W_ih = (const float*)d_in[1];
    const float* W_hh = (const float*)d_in[2];
    const float* b_h  = (const float*)d_in[3];
    const float* W_hy = (const float*)d_in[4];
    const float* b_y  = (const float*)d_in[5];
    float* y = (float*)d_out;

    char* ws = (char*)d_ws;
    const size_t h16_bytes  = (size_t)RNN_S * RNN_B * RNN_H * 2;   // 64 MB
    const size_t whhT_bytes = (size_t)RNN_H * RNN_H * 2;           // 512 KB
    _Float16* h16  = (_Float16*)ws;
    _Float16* WhhT = (_Float16*)(ws + h16_bytes);
    _Float16* WhyT = (_Float16*)(ws + h16_bytes + whhT_bytes);

    hipLaunchKernelGGL(rnn_prep_weights, dim3(1024), dim3(256), 0, stream,
                       W_hh, W_hy, WhhT, WhyT);
    hipLaunchKernelGGL(rnn_recurrence, dim3(4), dim3(512), 0, stream,
                       x, W_ih, b_h, WhhT, h16);
    hipLaunchKernelGGL(rnn_output, dim3(2048), dim3(256), 0, stream,
                       h16, WhyT, b_y, y);
}
